// CuboidTransformerAuxModel_51376398795117
// MI455X (gfx1250) — matrix operand order repaired and checked by reading, not compiled, not run on hardware
//
#include <hip/hip_runtime.h>
#include <cmath>

typedef __bf16 bf16_t;
typedef __attribute__((ext_vector_type(16))) __bf16 v16bf;
typedef __attribute__((ext_vector_type(8)))  __bf16 v8bf;
typedef __attribute__((ext_vector_type(8)))  float  v8f;
typedef __attribute__((ext_vector_type(4)))  unsigned int u32x4;
typedef __attribute__((ext_vector_type(4)))  int i32x4;
typedef __attribute__((ext_vector_type(8)))  int i32x8;

#define OUT_F32  0
#define OUT_BF16 1
#define OUT_ADD  2

#if defined(__gfx1250__) && __has_builtin(__builtin_amdgcn_tensor_load_to_lds)
#define HAVE_TDM 1
#else
#define HAVE_TDM 0
#endif

// ---------------- device helpers ----------------

__device__ __forceinline__ bf16_t f2bf(float f) {
  union { float f; unsigned u; } a; a.f = f;
  unsigned r = a.u + 0x7FFFu + ((a.u >> 16) & 1u);   // round to nearest even
  unsigned short h = (unsigned short)(r >> 16);
  bf16_t b; __builtin_memcpy(&b, &h, 2); return b;
}

__device__ __forceinline__ v16bf combine8(v8bf lo, v8bf hi) {
  v16bf r;
#pragma unroll
  for (int i = 0; i < 8; i++) { r[i] = lo[i]; r[8 + i] = hi[i]; }
  return r;
}

// A-matrix per-lane layout (16-bit, 16x32): two contiguous 8-elem chunks at
// k + (lane<16?0:8) and k+16 + (lane<16?0:8)
__device__ __forceinline__ v16bf load_a16(const bf16_t* row, int k, int lane) {
  int kh = (lane & 16) ? 8 : 0;
  v8bf lo = *(const v8bf*)(row + k + kh);
  v8bf hi = *(const v8bf*)(row + k + 16 + kh);
  return combine8(lo, hi);
}
// B-matrix per-lane layout (16-bit, 32x16): column n = lane%16, two contiguous
// 8-elem chunks at k + (lane<16?0:8) and k+16 + (lane<16?0:8); sourced from a [N,K] row-major buffer.
__device__ __forceinline__ v16bf load_b16(const bf16_t* col, int k, int lane) {
  int kh = (lane & 16) ? 8 : 0;
  v8bf lo = *(const v8bf*)(col + k + kh);
  v8bf hi = *(const v8bf*)(col + k + 16 + kh);
  return combine8(lo, hi);
}

// relative-position index inside a 4x4x4 cuboid (table is 7*7*7=343 entries)
__device__ __forceinline__ int rel_idx64(int m, int n) {
  int dt = (m >> 4) - (n >> 4) + 3;
  int dy = ((m >> 2) & 3) - ((n >> 2) & 3) + 3;
  int dx = (m & 3) - (n & 3) + 3;
  return (dt * 7 + dy) * 7 + dx;
}

// windowed-row -> original-row bijection.
// mode 0: identity, 1: local cuboid (4x4x4), 2: dilated cuboid, 3: cross (T x 4 x 4)
__device__ __forceinline__ long map_row(int row, int mode, int T, int H, int W,
                                        int st, int sh, int sw) {
  if (mode == 0) return row;
  int b, t, h, w;
  if (mode == 3) {
    int L = T * 16;
    int win = row / L, l = row - win * L;
    int nh = H >> 2, nw = W >> 2;
    int bb = win / (nh * nw), r2 = win - bb * (nh * nw);
    int ih = r2 / nw, iw = r2 - ih * nw;
    b = bb; t = l >> 4; h = ih * 4 + ((l >> 2) & 3); w = iw * 4 + (l & 3);
  } else {
    int win = row >> 6, l = row & 63;
    int lt = l >> 4, lh = (l >> 2) & 3, lw = l & 3;
    int nt = T >> 2, nh = H >> 2, nw = W >> 2;
    int bb = win / (nt * nh * nw), r2 = win - bb * (nt * nh * nw);
    int it = r2 / (nh * nw); r2 -= it * (nh * nw);
    int ih = r2 / nw, iw = r2 - ih * nw;
    b = bb;
    if (mode == 1) { t = it * 4 + lt; h = ih * 4 + lh; w = iw * 4 + lw; }
    else           { t = lt * nt + it; h = lh * nh + ih; w = lw * nw + iw; }
  }
  t += st; if (t >= T) t -= T;
  h += sh; if (h >= H) h -= H;
  w += sw; if (w >= W) w -= W;
  return (((long)b * T + t) * H + h) * W + w;
}

#if HAVE_TDM
// low 32 bits of a generic LDS pointer == LDS byte address (ISA: LDS aperture
// maps addr[31:0] to LDS)
__device__ __forceinline__ unsigned lds_off(const void* p) {
  return (unsigned)(size_t)p;
}
// Issue one TDM 2D tile load (bf16 elements): tile0 x tile1 elements from a
// row-major tensor with row stride stride0 (elements) into LDS at lds_addr.
// D# bitfields per CDNA5 ISA 8.3/8.4 (group0: count/lds/global/type=image;
// group1: data_size=2B + dims/tiles/strides). Groups 2/3 NULL (2D tensor).
__device__ __forceinline__ void tdm_load_2d(const bf16_t* gptr, unsigned lds_addr,
                                            unsigned tile0, unsigned tile1,
                                            unsigned dim0, unsigned dim1,
                                            unsigned long stride0) {
  unsigned long ga = (unsigned long)(size_t)gptr;
  u32x4 g0;
  g0[0] = 1u;                                                   // count=1, user D#
  g0[1] = lds_addr;                                             // lds_addr[31:0]
  g0[2] = (unsigned)ga;                                         // global_addr[31:0]
  g0[3] = (unsigned)((ga >> 32) & 0x01FFFFFFu) | 0x80000000u;   // addr[56:32] | type=2
  i32x8 g1;
  g1[0] = (int)(1u << 16);                                      // data_size=1 (2 bytes)
  g1[1] = (int)((dim0 & 0xFFFFu) << 16);                        // tensor_dim0 lo
  g1[2] = (int)((dim0 >> 16) | ((dim1 & 0xFFFFu) << 16));       // dim0 hi | dim1 lo
  g1[3] = (int)((dim1 >> 16) | (tile0 << 16));                  // dim1 hi | tile_dim0
  g1[4] = (int)(tile1 & 0xFFFFu);                               // tile_dim1 (tile2=0)
  g1[5] = (int)(unsigned)stride0;                               // dim0_stride[31:0]
  g1[6] = (int)(unsigned)((stride0 >> 32) & 0xFFFFu);           // dim0_stride[47:32]
  g1[7] = 0;                                                    // dim1_stride unused
  i32x4 z4 = {};
#if __has_include(<hip/amd_detail/amd_gfx1250_TDM.h>)
  i32x8 z8 = {};
  __builtin_amdgcn_tensor_load_to_lds(g0, g1, z4, z4, z8, 0);   // 6-arg toolchain
#else
  __builtin_amdgcn_tensor_load_to_lds(g0, g1, z4, z4, 0);       // 5-arg ROCm 7.2
#endif
}
__device__ __forceinline__ void wait_tensor0() {
#if __has_builtin(__builtin_amdgcn_s_wait_tensorcnt)
  __builtin_amdgcn_s_wait_tensorcnt(0);
#endif
}
#endif  // HAVE_TDM

// cooperative fallback tile stage (rows x 32 bf16, LDS row stride 32)
__device__ __forceinline__ void stage_tile(bf16_t* dst, const bf16_t* src,
                                           int rows, int ldsrc, int tid) {
  for (int i = tid; i < rows * 2; i += 256) {
    int r = i >> 1, half = (i & 1) * 16;
    *(v16bf*)(dst + r * 32 + half) = *(const v16bf*)(src + (size_t)r * ldsrc + half);
  }
}

// ---------------- kernels ----------------

// f32 [K,N] weight -> bf16 [N,K] (transposed, WMMA-B friendly)
__global__ void wtrans_kernel(const float* __restrict__ W, bf16_t* __restrict__ WT,
                              int K, int N) {
  long i = (long)blockIdx.x * 256 + threadIdx.x;
  if (i >= (long)K * N) return;
  int n = (int)(i % N), k = (int)(i / N);
  WT[(long)n * K + k] = f2bf(W[i]);
}

// fused LayerNorm + roll + window gather, bf16 output (one block per output row)
__global__ __launch_bounds__(128)
void ln_gather_kernel(const float* __restrict__ x, const float* __restrict__ gam,
                      const float* __restrict__ bet, bf16_t* __restrict__ out,
                      int rows, int C, int mode, int T, int H, int W,
                      int st, int sh, int sw) {
  __shared__ float red[128];
  int row = blockIdx.x;
  long src = map_row(row, mode, T, H, W, st, sh, sw);
  const float* xr = x + src * (long)C;
  int tid = threadIdx.x;
  float s = 0.f;
  for (int c = tid; c < C; c += 128) s += xr[c];
  red[tid] = s; __syncthreads();
  for (int k = 64; k > 0; k >>= 1) { if (tid < k) red[tid] += red[tid + k]; __syncthreads(); }
  float mean = red[0] / C; __syncthreads();
  float vs = 0.f;
  for (int c = tid; c < C; c += 128) { float d = xr[c] - mean; vs += d * d; }
  red[tid] = vs; __syncthreads();
  for (int k = 64; k > 0; k >>= 1) { if (tid < k) red[tid] += red[tid + k]; __syncthreads(); }
  float rstd = rsqrtf(red[0] / C + 1e-5f);
  bf16_t* orow = out + (long)row * C;
  for (int c = tid; c < C; c += 128)
    orow[c] = f2bf((xr[c] - mean) * rstd * gam[c] + bet[c]);
}

// windowed f32 rows scatter-added back (window-reverse + unroll + residual)
__global__ void scatter_add_kernel(const float* __restrict__ src, float* __restrict__ dst,
                                   int rows, int C, int mode, int T, int H, int W,
                                   int st, int sh, int sw) {
  long i = (long)blockIdx.x * 256 + threadIdx.x;
  if (i >= (long)rows * C) return;
  int row = (int)(i / C), c = (int)(i % C);
  long d = map_row(row, mode, T, H, W, st, sh, sw);
  dst[d * C + c] += src[i];
}

// [B,T,H,W,C] -> [B,T,H/2,W/2,4C] with c4 = (i*2+j)*C + c
__global__ void patch_reorder_kernel(const float* __restrict__ x, float* __restrict__ out,
                                     int B, int T, int H, int W, int C) {
  long i = (long)blockIdx.x * 256 + threadIdx.x;
  if (i >= (long)B * T * H * W * C) return;
  int c = (int)(i % C); long r = i / C;
  int w = (int)(r % W); r /= W;
  int h = (int)(r % H); r /= H;           // r = b*T + t
  int h2 = h >> 1, ii = h & 1, w2 = w >> 1, jj = w & 1;
  long orow = (r * (H / 2) + h2) * (W / 2) + w2;
  out[orow * (4L * C) + (long)(ii * 2 + jj) * C + c] = x[i];
}

// nearest-2x upsample + 3x3 SAME conv (NHWC, weights [3,3,Ci,Co] f32)
__global__ __launch_bounds__(256)
void upsample_conv_kernel(const float* __restrict__ x, const float* __restrict__ w,
                          const float* __restrict__ b, float* __restrict__ out,
                          int N, int Ho, int Wo, int Ci, int Co) {
  long i = (long)blockIdx.x * 256 + threadIdx.x;
  if (i >= (long)N * Ho * Wo * Co) return;
  int co = (int)(i % Co); long r = i / Co;
  int ox = (int)(r % Wo); r /= Wo;
  int oy = (int)(r % Ho); r /= Ho;
  int n = (int)r;
  int Hi = Ho >> 1, Wi = Wo >> 1;
  float acc = b[co];
  for (int dy = 0; dy < 3; dy++) {
    int uy = oy + dy - 1; if (uy < 0 || uy >= Ho) continue;
    int iy = uy >> 1;
    for (int dx = 0; dx < 3; dx++) {
      int ux = ox + dx - 1; if (ux < 0 || ux >= Wo) continue;
      int ix = ux >> 1;
      const float* xp = x + (((long)n * Hi + iy) * Wi + ix) * Ci;
      const float* wp = w + (long)(dy * 3 + dx) * Ci * Co + co;
      for (int c = 0; c < Ci; c++) acc += xp[c] * wp[(long)c * Co];
    }
  }
  out[i] = acc;
}

#define GEMM_STEP(as, bs)                                                                                \
  do {                                                                                                   \
    v16bf a0 = load_a16((as) + (wm * 32 + lrow) * 32, 0, lane);                                          \
    v16bf a1 = load_a16((as) + (wm * 32 + 16 + lrow) * 32, 0, lane);                                     \
    v16bf b0 = load_b16((bs) + (wn * 32 + lrow) * 32, 0, lane);                                          \
    v16bf b1 = load_b16((bs) + (wn * 32 + 16 + lrow) * 32, 0, lane);                                     \
    acc00 = __builtin_amdgcn_wmma_f32_16x16x32_bf16(false, a0, false, b0, (short)0, acc00, false, false);\
    acc01 = __builtin_amdgcn_wmma_f32_16x16x32_bf16(false, a0, false, b1, (short)0, acc01, false, false);\
    acc10 = __builtin_amdgcn_wmma_f32_16x16x32_bf16(false, a1, false, b0, (short)0, acc10, false, false);\
    acc11 = __builtin_amdgcn_wmma_f32_16x16x32_bf16(false, a1, false, b1, (short)0, acc11, false, false);\
  } while (0)

// bf16 WMMA GEMM: out[M,N] = A[M,K] * BT[N,K]^T + bias, fused epilogue.
// 256 threads = 8 waves (2x4), each wave a 32x32 tile. A/B K-slabs are
// DMA'd into LDS by the Tensor Data Mover (double-buffered, TENSORcnt-waited),
// so WMMA operands stream from LDS instead of 8x-redundant global loads.
template <int OUTMODE, bool LEAKY>
__global__ __launch_bounds__(256)
void gemm_bf16_kernel(const bf16_t* __restrict__ A, const bf16_t* __restrict__ BT,
                      const float* __restrict__ bias, void* __restrict__ outp,
                      int M, int N, int K) {
  __shared__ __align__(32) bf16_t As[2][64 * 32];    // 8 KB
  __shared__ __align__(32) bf16_t Bs[2][128 * 32];   // 16 KB
  int tid = threadIdx.x;
  int wave = tid >> 5, lane = tid & 31, lrow = lane & 15;
  int wm = wave & 1, wn = wave >> 1;
  int mblock = blockIdx.x * 64, nblock = blockIdx.y * 128;
  const bf16_t* Ab = A + (size_t)mblock * K;
  const bf16_t* Bb = BT + (size_t)nblock * K;
  v8f acc00 = {}, acc01 = {}, acc10 = {}, acc11 = {};

#if HAVE_TDM
  if (tid == 0) {
    tdm_load_2d(Ab, lds_off(&As[0][0]), 32, 64, (unsigned)K, 64u, (unsigned long)K);
    tdm_load_2d(Bb, lds_off(&Bs[0][0]), 32, 128, (unsigned)K, 128u, (unsigned long)K);
  }
  wait_tensor0();
  __syncthreads();
  int cur = 0;
  for (int k = 0; k < K; k += 32) {
    if (k + 32 < K && tid == 0) {   // prefetch next K-slab while computing
      tdm_load_2d(Ab + k + 32, lds_off(&As[cur ^ 1][0]), 32, 64, (unsigned)K, 64u, (unsigned long)K);
      tdm_load_2d(Bb + k + 32, lds_off(&Bs[cur ^ 1][0]), 32, 128, (unsigned)K, 128u, (unsigned long)K);
    }
    GEMM_STEP(As[cur], Bs[cur]);
    wait_tensor0();                 // wave 0 drains its DMA; others pass through
    __syncthreads();
    cur ^= 1;
  }
#else
  for (int k = 0; k < K; k += 32) {
    stage_tile(As[0], Ab + k, 64, K, tid);
    stage_tile(Bs[0], Bb + k, 128, K, tid);
    __syncthreads();
    GEMM_STEP(As[0], Bs[0]);
    __syncthreads();
  }
#endif

  int mo = (lane & 16) ? 8 : 0;
  v8f accs[4] = {acc00, acc01, acc10, acc11};
#pragma unroll
  for (int t = 0; t < 4; t++) {
    int mt = t >> 1, nt = t & 1;
    int n = nblock + wn * 32 + nt * 16 + lrow;
    float bv = bias ? bias[n] : 0.0f;
#pragma unroll
    for (int r = 0; r < 8; r++) {
      int m = mblock + wm * 32 + mt * 16 + r + mo;
      float v = accs[t][r] + bv;
      if (LEAKY) v = (v >= 0.0f) ? v : 0.1f * v;
      size_t off = (size_t)m * N + n;
      if (OUTMODE == OUT_BF16)      ((bf16_t*)outp)[off] = f2bf(v);
      else if (OUTMODE == OUT_F32)  ((float*)outp)[off] = v;
      else                          ((float*)outp)[off] += v;
    }
  }
}

// windowed attention: S = scale*Q*K^T (+rel bias), softmax, O = P*V.
// grid (nWin, heads, Lq/32), 128 threads (4 waves). All matmuls via WMMA bf16.
__global__ __launch_bounds__(128)
void attn_kernel(const bf16_t* __restrict__ Q, const bf16_t* __restrict__ Kp,
                 const bf16_t* __restrict__ Vp, bf16_t* __restrict__ O,
                 int Lq, int Lk, int dh, int qld, int kld, int oldd,
                 int qoff, int koff, int voff, int ooff,
                 float scale, const float* __restrict__ rel_bias) {
  __shared__ __align__(32) bf16_t Vt[64 * 128];   // V^T staged: [dh][Lk]
  __shared__ __align__(32) float  Sf[32 * 128];   // logits chunk [32][Lk]
  __shared__ __align__(32) bf16_t Ps[32 * 128];   // probs bf16  [32][Lk]
  int win = blockIdx.x, head = blockIdx.y, mchunk = blockIdx.z;
  int tid = threadIdx.x, wave = tid >> 5, lane = tid & 31, lrow = lane & 15;
  long qbase = (long)win * Lq, kbase = (long)win * Lk;
  int hq = qoff + head * dh, hk = koff + head * dh, hv = voff + head * dh;

  for (int i = tid; i < dh * Lk; i += 128) {
    int d = i / Lk, t = i - d * Lk;
    Vt[d * Lk + t] = Vp[(kbase + t) * kld + hv + d];
  }
  __syncthreads();

  int ntiles = Lk >> 4;
  int mo = (lane & 16) ? 8 : 0;
  for (int tile = wave; tile < 2 * ntiles; tile += 4) {
    int mt = tile & 1, nt = tile >> 1;
    const bf16_t* qrow = Q + (qbase + mchunk * 32 + mt * 16 + lrow) * qld + hq;
    const bf16_t* krow = Kp + (kbase + nt * 16 + lrow) * kld + hk;
    v8f acc = {};
    for (int kk = 0; kk < dh; kk += 32) {
      v16bf a = load_a16(qrow, kk, lane);
      v16bf b = load_b16(krow, kk, lane);
      acc = __builtin_amdgcn_wmma_f32_16x16x32_bf16(false, a, false, b, (short)0, acc, false, false);
    }
    int n = nt * 16 + lrow;
#pragma unroll
    for (int r = 0; r < 8; r++) {
      int lm = mt * 16 + r + mo;
      float v = acc[r] * scale;
      if (rel_bias) v += rel_bias[rel_idx64(mchunk * 32 + lm, n) * 4 + head];
      Sf[lm * Lk + n] = v;
    }
  }
  __syncthreads();

  if (tid < 32) {
    float mx = -1e30f;
    for (int j = 0; j < Lk; j++) mx = fmaxf(mx, Sf[tid * Lk + j]);
    float s = 0.f;
    for (int j = 0; j < Lk; j++) { float e = __expf(Sf[tid * Lk + j] - mx); Sf[tid * Lk + j] = e; s += e; }
    float inv = 1.0f / s;
    for (int j = 0; j < Lk; j++) Ps[tid * Lk + j] = f2bf(Sf[tid * Lk + j] * inv);
  }
  __syncthreads();

  int ntO = dh >> 4;
  for (int tile = wave; tile < 2 * ntO; tile += 4) {
    int mt = tile & 1, nt = tile >> 1;
    const bf16_t* prow = Ps + (mt * 16 + lrow) * Lk;
    const bf16_t* vrow = Vt + (nt * 16 + lrow) * Lk;
    v8f acc = {};
    for (int kk = 0; kk < Lk; kk += 32) {
      v16bf a = load_a16(prow, kk, lane);
      v16bf b = load_b16(vrow, kk, lane);
      acc = __builtin_amdgcn_wmma_f32_16x16x32_bf16(false, a, false, b, (short)0, acc, false, false);
    }
#pragma unroll
    for (int r = 0; r < 8; r++) {
      long orow = qbase + mchunk * 32 + mt * 16 + r + mo;
      O[orow * oldd + ooff + head * dh + nt * 16 + lrow] = f2bf(acc[r]);
    }
  }
}

// ---------------- host orchestration ----------------

struct AttnIdx { int ln_s, ln_b, w_qkv, b_qkv, w_proj, b_proj, rel; };
static AttnIdx attn_at(int b) { return {b, b + 1, b + 2, b + 3, b + 4, b + 5, b + 6}; }
struct CrossIdx { int ln_q_s, ln_q_b, ln_kv_s, ln_kv_b, w_q, b_q, w_kv, b_kv, w_proj, b_proj; };
static CrossIdx cross_at(int b) { return {b, b + 1, b + 2, b + 3, b + 4, b + 5, b + 6, b + 7, b + 8, b + 9}; }
struct FfnIdx { int ln_s, ln_b, w1, b1, w2, b2; };
static FfnIdx ffn_at(int b) { return {b, b + 1, b + 2, b + 3, b + 4, b + 5}; }

extern "C" void kernel_launch(void* const* d_in, const int* in_sizes, int n_in,
                              void* d_out, int out_size, void* d_ws, size_t ws_size,
                              hipStream_t stream) {
  (void)in_sizes; (void)n_in; (void)out_size; (void)ws_size;
  auto F = [&](int i) { return (const float*)d_in[i]; };

  char* ws = (char*)d_ws;
  bf16_t* WT   = (bf16_t*)(ws);                      // 16 MB  transposed bf16 weights
  bf16_t* LNB  = (bf16_t*)(ws + (16ull  << 20));     // 36 MB  LN/gather output (GEMM A)
  bf16_t* ATO  = (bf16_t*)(ws + (52ull  << 20));     // 36 MB  attn out / kv-LN
  bf16_t* BIG  = (bf16_t*)(ws + (88ull  << 20));     // 136 MB qkv / ffn hidden / q-proj
  bf16_t* BIG2 = (bf16_t*)(ws + (156ull << 20));     //        kv-proj half of BIG
  float*  TMP  = (float*) (ws + (224ull << 20));     // 68 MB  f32 windowed proj output
  float*  ACT0 = (float*) (ws + (292ull << 20));     // 68 MB  level-0 activation
  float*  ACT1 = (float*) (ws + (360ull << 20));     // 68 MB  level-1 activation

  size_t wtoff = 0;
  auto trans = [&](int idx, int K, int N) -> const bf16_t* {
    bf16_t* dst = WT + wtoff;
    long total = (long)K * N;
    wtrans_kernel<<<(int)((total + 255) / 256), 256, 0, stream>>>(F(idx), dst, K, N);
    wtoff += (size_t)total;
    return dst;
  };
  auto gemm = [&](const bf16_t* A, const bf16_t* BT, const float* bias, void* outp,
                  int M, int N, int K, int mode, bool leaky) {
    dim3 g(M / 64, N / 128);
    if (mode == OUT_BF16 && leaky)
      gemm_bf16_kernel<OUT_BF16, true><<<g, 256, 0, stream>>>(A, BT, bias, outp, M, N, K);
    else if (mode == OUT_BF16)
      gemm_bf16_kernel<OUT_BF16, false><<<g, 256, 0, stream>>>(A, BT, bias, outp, M, N, K);
    else if (mode == OUT_F32)
      gemm_bf16_kernel<OUT_F32, false><<<g, 256, 0, stream>>>(A, BT, bias, outp, M, N, K);
    else
      gemm_bf16_kernel<OUT_ADD, false><<<g, 256, 0, stream>>>(A, BT, bias, outp, M, N, K);
  };

  auto self_attn = [&](float* act, int T, int H, int W, int C, AttnIdx ai, int mode, int sft) {
    int rows = 4 * T * H * W, nW = rows / 64, dh = C / 4;
    ln_gather_kernel<<<rows, 128, 0, stream>>>(act, F(ai.ln_s), F(ai.ln_b), LNB,
                                               rows, C, mode, T, H, W, sft, sft, sft);
    const bf16_t* WQKV = trans(ai.w_qkv, C, 3 * C);
    gemm(LNB, WQKV, F(ai.b_qkv), BIG, rows, 3 * C, C, OUT_BF16, false);
    attn_kernel<<<dim3(nW, 4, 2), 128, 0, stream>>>(BIG, BIG, BIG, ATO, 64, 64, dh,
        3 * C, 3 * C, C, 0, C, 2 * C, 0, 1.0f / sqrtf((float)dh), F(ai.rel));
    const bf16_t* WP = trans(ai.w_proj, C, C);
    gemm(ATO, WP, F(ai.b_proj), TMP, rows, C, C, OUT_F32, false);
    long total = (long)rows * C;
    scatter_add_kernel<<<(int)((total + 255) / 256), 256, 0, stream>>>(TMP, act, rows, C,
                                                                      mode, T, H, W, sft, sft, sft);
  };

  auto cross_attn = [&](float* act, const float* mem, int T, int H, int W, int C, CrossIdx ci) {
    int rows = 4 * T * H * W;
    int nW = 4 * (H / 4) * (W / 4);
    int Lq = T * 16, Lk = T * 16, dh = C / 4;
    ln_gather_kernel<<<rows, 128, 0, stream>>>(act, F(ci.ln_q_s), F(ci.ln_q_b), LNB,
                                               rows, C, 3, T, H, W, 0, 0, 0);
    ln_gather_kernel<<<rows, 128, 0, stream>>>(mem, F(ci.ln_kv_s), F(ci.ln_kv_b), ATO,
                                               rows, C, 3, T, H, W, 0, 0, 0);
    const bf16_t* WQ = trans(ci.w_q, C, C);
    gemm(LNB, WQ, F(ci.b_q), BIG, rows, C, C, OUT_BF16, false);
    const bf16_t* WKV = trans(ci.w_kv, C, 2 * C);
    gemm(ATO, WKV, F(ci.b_kv), BIG2, rows, 2 * C, C, OUT_BF16, false);
    attn_kernel<<<dim3(nW, 4, Lq / 32), 128, 0, stream>>>(BIG, BIG2, BIG2, ATO, Lq, Lk, dh,
        C, 2 * C, C, 0, 0, C, 0, 1.0f / sqrtf((float)dh), nullptr);
    const bf16_t* WP = trans(ci.w_proj, C, C);
    gemm(ATO, WP, F(ci.b_proj), TMP, rows, C, C, OUT_F32, false);
    long total = (long)rows * C;
    scatter_add_kernel<<<(int)((total + 255) / 256), 256, 0, stream>>>(TMP, act, rows, C,
                                                                      3, T, H, W, 0, 0, 0);
  };

  auto ffn = [&](float* act, float* dst, int rows, int C, FfnIdx fi) {
    ln_gather_kernel<<<rows, 128, 0, stream>>>(act, F(fi.ln_s), F(fi.ln_b), LNB,
                                               rows, C, 0, 1, 1, 1, 0, 0, 0);
    const bf16_t* W1 = trans(fi.w1, C, 4 * C);
    gemm(LNB, W1, F(fi.b1), BIG, rows, 4 * C, C, OUT_BF16, true);   // fused leaky-relu
    const bf16_t* W2 = trans(fi.w2, 4 * C, C);
    gemm(BIG, W2, F(fi.b2), dst, rows, C, 4 * C, OUT_ADD, false);   // fused residual
  };

  const float* X    = (const float*)d_in[0];
  const float* MEM0 = (const float*)d_in[1];
  const float* MEM1 = (const float*)d_in[2];
  float* OUT = (float*)d_out;

  // ---- down0 (C=128, 8x64x64) ----
  hipMemcpyAsync(ACT0, X, (size_t)131072 * 128 * 4, hipMemcpyDeviceToDevice, stream);
  self_attn(ACT0, 8, 64, 64, 128, attn_at(3), 1, 1);
  self_attn(ACT0, 8, 64, 64, 128, attn_at(10), 2, 0);
  ffn(ACT0, ACT0, 131072, 128, ffn_at(17));

  // ---- patch merge -> level1 (C=256, 8x32x32) ----
  {
    long total = (long)131072 * 128;
    patch_reorder_kernel<<<(int)((total + 255) / 256), 256, 0, stream>>>(ACT0, TMP, 4, 8, 64, 64, 128);
    ln_gather_kernel<<<32768, 128, 0, stream>>>(TMP, F(23), F(24), LNB, 32768, 512, 0, 1, 1, 1, 0, 0, 0);
    const bf16_t* WPM = trans(25, 512, 256);
    gemm(LNB, WPM, F(26), ACT1, 32768, 256, 512, OUT_F32, false);
  }

  // ---- down1 ----
  self_attn(ACT1, 8, 32, 32, 256, attn_at(27), 1, 1);
  self_attn(ACT1, 8, 32, 32, 256, attn_at(34), 2, 0);
  ffn(ACT1, ACT1, 32768, 256, ffn_at(41));

  // ---- up1 (self + 2x cross to mem1) ----
  self_attn(ACT1, 8, 32, 32, 256, attn_at(47), 1, 1);
  self_attn(ACT1, 8, 32, 32, 256, attn_at(54), 2, 0);
  cross_attn(ACT1, MEM1, 8, 32, 32, 256, cross_at(61));
  cross_attn(ACT1, MEM1, 8, 32, 32, 256, cross_at(71));
  ffn(ACT1, ACT1, 32768, 256, ffn_at(81));

  // ---- upsample (nearest 2x + 3x3 conv, 256 -> 128) ----
  {
    long total = (long)32 * 64 * 64 * 128;
    upsample_conv_kernel<<<(int)((total + 255) / 256), 256, 0, stream>>>(
        ACT1, F(87), F(88), ACT0, 32, 64, 64, 256, 128);
  }

  // ---- up0 (self + 2x cross to mem0) ----
  self_attn(ACT0, 8, 64, 64, 128, attn_at(89), 1, 1);
  self_attn(ACT0, 8, 64, 64, 128, attn_at(96), 2, 0);
  cross_attn(ACT0, MEM0, 8, 64, 64, 128, cross_at(103));
  cross_attn(ACT0, MEM0, 8, 64, 64, 128, cross_at(113));

  // final FFN: residual accumulated straight into d_out
  hipMemcpyAsync(OUT, ACT0, (size_t)131072 * 128 * 4, hipMemcpyDeviceToDevice, stream);
  ffn(ACT0, OUT, 131072, 128, ffn_at(123));
}
